// NextVLAD_33913061769333
// MI455X (gfx1250) — matrix operand order repaired and checked by reading, not compiled, not run on hardware
//
#include <hip/hip_runtime.h>

// ---------------------------------------------------------------------------
// NextVLAD + MPNCOV (Newton-Schulz sqrtm) for MI455X / gfx1250, wave32 WMMA.
// Big GEMMs: TDM (tensor_load_to_lds) double-buffered 128x128 block tiles,
// 8 waves x (32x64 per wave) register blocking, v_wmma_f32_16x16x32_bf16.
// NS matmuls: v_wmma_f32_16x16x4_f32. Everything else f32 VALU.
// Workspace use: ~116 MB.
// ---------------------------------------------------------------------------

typedef __bf16 bf16;
typedef __attribute__((ext_vector_type(16))) __bf16 bf16x16;
typedef __attribute__((ext_vector_type(8)))  __bf16 bf16x8;
typedef __attribute__((ext_vector_type(8)))  float  f32x8;
typedef __attribute__((ext_vector_type(2)))  float  f32x2;
typedef __attribute__((ext_vector_type(4)))  unsigned int u32x4;
typedef __attribute__((ext_vector_type(8)))  int i32x8;
typedef __attribute__((ext_vector_type(4)))  int i32x4;

#define BS   4        // batches
#define MTK  1024     // tokens per batch (M)
#define NF   1024     // feature dim
#define EN   2048     // expanded dim E*DIM
#define GG   8        // groups
#define KCL  128      // clusters
#define DGD  256      // per-group dim
#define GKN  1024     // G*K
#define DD   256      // cov dim
#define TRIU 32896    // DD*(DD+1)/2

#if __has_builtin(__builtin_amdgcn_tensor_load_to_lds)
#define NV_HAVE_TDM 1
#else
#define NV_HAVE_TDM 0
#endif

// A-operand fragment: lane holds row (lane&15); 16 bf16 = K chunks
// [hf*8, hf*8+8) and [16+hf*8, 16+hf*8+8)  (ISA 16-bit A 16x32 table).
// Caller pre-offsets pointer by row*lda + hf*8; chunks are +0 and +16.
__device__ __forceinline__ bf16x16 load_frag_a(const bf16* p) {
  bf16x8 lo = *(const bf16x8*)p;
  bf16x8 hi = *(const bf16x8*)(p + 16);
  bf16x16 r;
#pragma unroll
  for (int i = 0; i < 8; ++i) { r[i] = lo[i]; r[i + 8] = hi[i]; }
  return r;
}

// B-operand fragment: lane holds column n=(lane&15); K chunks
// [hf*8, hf*8+8) and [16+hf*8, 16+hf*8+8). Caller pre-offsets by n*ldb + hf*8.
__device__ __forceinline__ bf16x16 load_frag_b(const bf16* p) {
  bf16x8 lo = *(const bf16x8*)p;
  bf16x8 hi = *(const bf16x8*)(p + 16);
  bf16x16 r;
#pragma unroll
  for (int i = 0; i < 8; ++i) { r[i] = lo[i]; r[i + 8] = hi[i]; }
  return r;
}

#if NV_HAVE_TDM
// Issue one TDM 2D tile load: rows x cols bf16, row stride = stride_elems,
// into LDS at lds_dst (contiguous rows*cols bf16). D# per ISA 8.3/8.4.
// 6-arg builtin variant (clang-23 / therock): (g0, g1, g2, g3, g_extra, cpol).
__device__ __forceinline__ void tdm_load_tile_bf16(bf16* lds_dst, const bf16* gsrc,
                                                   int rows, int cols, int stride_elems) {
  unsigned lds_off = (unsigned)(unsigned long long)lds_dst;  // LDS aperture: addr[31:0]
  unsigned long long ga = (unsigned long long)gsrc;
  u32x4 g0;
  g0[0] = 1u;                                            // count=1, user mode
  g0[1] = lds_off;                                       // lds_addr (bytes)
  g0[2] = (unsigned)ga;                                  // global_addr[31:0]
  g0[3] = (unsigned)((ga >> 32) & 0x01ffffffull) | 0x80000000u;  // addr[56:32], type=2
  unsigned td0 = (unsigned)stride_elems;                 // tensor_dim0 (>= cols, no OOB)
  unsigned td1 = (unsigned)rows;                         // tensor_dim1
  i32x8 g1;
  g1[0] = 0x10000;                                       // wg_mask=0, data_size=1 (2B)
  g1[1] = (int)(td0 << 16);                              // tensor_dim0[15:0]
  g1[2] = (int)((td0 >> 16) | (td1 << 16));              // tensor_dim0[31:16] | dim1[15:0]
  g1[3] = (int)((td1 >> 16) | ((unsigned)cols << 16));   // dim1[31:16] | tile_dim0
  g1[4] = (int)rows;                                     // tile_dim1; tile_dim2=0
  g1[5] = (int)stride_elems;                             // tensor_dim0_stride[31:0]
  g1[6] = 0;                                             // stride[47:32] | dim1_stride lo
  g1[7] = 0;
  i32x4 z4 = {0, 0, 0, 0};
  i32x8 z8 = {0, 0, 0, 0, 0, 0, 0, 0};
  __builtin_amdgcn_tensor_load_to_lds(g0, g1, z4, z4, z8, 0);
}
#else
// Fallback: cooperative copy, all 256 threads. 128x64 bf16 tile.
__device__ __forceinline__ void copy_tile_all(bf16* lds_dst, const bf16* gsrc,
                                              int stride_elems) {
#pragma unroll
  for (int q = 0; q < 4; ++q) {
    int c = threadIdx.x + 256 * q;        // 1024 chunks of 8 bf16
    int rr = c >> 3, c8 = (c & 7) * 8;
    *(bf16x8*)(lds_dst + rr * 64 + c8) = *(const bf16x8*)(gsrc + (size_t)rr * stride_elems + c8);
  }
}
#endif

// ---------------------------------------------------------------------------
// Stage 0: casts / layout prep
// ---------------------------------------------------------------------------
__global__ void nv_cast_bf16(const float* __restrict__ src, bf16* __restrict__ dst, int n) {
  int i = blockIdx.x * blockDim.x + threadIdx.x;
  if (i < n) dst[i] = (bf16)src[i];
}

// x[32,1024,16,8] -> xt[b,m,c] L2-normalized over c, bf16. Block per (b,m).
__global__ void nv_prep_normalize(const float* __restrict__ x, bf16* __restrict__ xtn) {
  int bm = blockIdx.x;                // b*1024 + m
  int b = bm >> 10, m = bm & 1023;
  int j = m >> 7, hw = m & 127;       // m = j*128 + h*8 + w
  const float* src = x + (size_t)(b * 8 + j) * 131072 + hw;  // stride 128 per c
  __shared__ float red[256];
  int t = threadIdx.x;
  float ss = 0.f;
  for (int c = t; c < NF; c += 256) { float v = src[(size_t)c * 128]; ss += v * v; }
  red[t] = ss; __syncthreads();
  for (int s = 128; s; s >>= 1) { if (t < s) red[t] += red[t + s]; __syncthreads(); }
  float inv = 1.0f / fmaxf(sqrtf(red[0]), 1e-12f);
  bf16* dst = xtn + (size_t)bm * NF;
  for (int c = t; c < NF; c += 256) dst[c] = (bf16)(src[(size_t)c * 128] * inv);
}

// ---------------------------------------------------------------------------
// TDM-staged bf16 WMMA GEMM: C[b, M, N] = A[b, M, K] @ B[N, K]^T  (+ epilogue)
// Block tile 128x128, double-buffered 128x64 LDS stages for A and B.
// 8 waves; each wave computes 32x64 (2x4 WMMA tiles, 8 f32x8 accumulators).
// mode 0: C = P + bias[n]     mode 1: C = P - wsum[b*Mrows+m]*cent[m*ldc+n]
// ---------------------------------------------------------------------------
__global__ void __launch_bounds__(256)
nv_gemm_tdm(const bf16* __restrict__ A, long sA,
            const bf16* __restrict__ B, long sB,
            float* __restrict__ C, long sC,
            int MB, int NB, int Kdim, int lda, int ldb, int ldc,
            const float* __restrict__ bias,
            const float* __restrict__ wsum, const float* __restrict__ cent,
            int Mrows, int mode, int batches) {
  __shared__ __align__(16) bf16 smA[2][128 * 64];
  __shared__ __align__(16) bf16 smB[2][128 * 64];
  int blk = blockIdx.x;
  int b = blk / (MB * NB);
  int r = blk - b * MB * NB;
  int tm = r / NB, tn = r - tm * NB;
  int bm = tm * 128, bn = tn * 128;
  int wid = threadIdx.x >> 5;
  int lane = threadIdx.x & 31;
  int row = lane & 15, hf = lane >> 4;
  int wm = wid >> 1, wn = wid & 1;    // wave grid: 4(M) x 2(N)

  const bf16* Ab = A + (size_t)b * sA + (size_t)bm * lda;
  const bf16* Bb = B + (size_t)b * sB + (size_t)bn * ldb;

  const f32x8 fz = {};
  f32x8 acc[2][4];
#pragma unroll
  for (int i = 0; i < 2; ++i)
#pragma unroll
    for (int j = 0; j < 4; ++j) acc[i][j] = fz;

#if NV_HAVE_TDM
  if (wid == 0) {
    tdm_load_tile_bf16(&smA[0][0], Ab, 128, 64, lda);
    tdm_load_tile_bf16(&smB[0][0], Bb, 128, 64, ldb);
    __builtin_amdgcn_s_wait_tensorcnt((short)0);
  }
#else
  copy_tile_all(&smA[0][0], Ab, lda);
  copy_tile_all(&smB[0][0], Bb, ldb);
#endif
  __syncthreads();

  int p = 0;
  for (int ks = 0; ks < Kdim; ks += 64) {
    int nxt = ks + 64;
    bool pf = (nxt < Kdim);
#if NV_HAVE_TDM
    if (wid == 0 && pf) {               // DMA next stage while computing
      tdm_load_tile_bf16(&smA[p ^ 1][0], Ab + nxt, 128, 64, lda);
      tdm_load_tile_bf16(&smB[p ^ 1][0], Bb + nxt, 128, 64, ldb);
    }
#else
    if (pf) {
      copy_tile_all(&smA[p ^ 1][0], Ab + nxt, lda);
      copy_tile_all(&smB[p ^ 1][0], Bb + nxt, ldb);
    }
#endif
#pragma unroll
    for (int s = 0; s < 2; ++s) {       // two K=32 sub-steps of the 64-wide stage
      bf16x16 af[2], bfr[4];
#pragma unroll
      for (int i = 0; i < 2; ++i)
        af[i] = load_frag_a(&smA[p][(wm * 32 + i * 16 + row) * 64 + s * 32 + hf * 8]);
#pragma unroll
      for (int j = 0; j < 4; ++j)
        bfr[j] = load_frag_b(&smB[p][(wn * 64 + j * 16 + row) * 64 + s * 32 + hf * 8]);
#pragma unroll
      for (int i = 0; i < 2; ++i)
#pragma unroll
        for (int j = 0; j < 4; ++j)
          acc[i][j] = __builtin_amdgcn_wmma_f32_16x16x32_bf16(
              false, af[i], false, bfr[j], (short)0, acc[i][j], false, false);
    }
#if NV_HAVE_TDM
    if (wid == 0 && pf) __builtin_amdgcn_s_wait_tensorcnt((short)0);
#endif
    __syncthreads();
    p ^= 1;
  }

  // epilogue
#pragma unroll
  for (int i = 0; i < 2; ++i) {
    int r0 = bm + wm * 32 + i * 16 + hf * 8;
#pragma unroll
    for (int j = 0; j < 4; ++j) {
      int cn = bn + wn * 64 + j * 16 + row;
      float* pc = C + (size_t)b * sC + (size_t)r0 * ldc + cn;
      if (mode == 0) {
        float bv = bias ? bias[cn] : 0.f;
#pragma unroll
        for (int q = 0; q < 8; ++q) pc[(size_t)q * ldc] = acc[i][j][q] + bv;
      } else {
        const float* wsb = wsum + (size_t)b * Mrows;
#pragma unroll
        for (int q = 0; q < 8; ++q) {
          int rr = r0 + q;
          pc[(size_t)q * ldc] = acc[i][j][q] - wsb[rr] * cent[(size_t)rr * ldc + cn];
        }
      }
    }
  }
}

// h f32 [b,m,e] -> bf16 same layout + bf16 transposed [b,e,m]
__global__ void nv_h_cast_tr(const float* __restrict__ h,
                             bf16* __restrict__ hbf, bf16* __restrict__ hbfT) {
  size_t i = (size_t)blockIdx.x * blockDim.x + threadIdx.x;   // 8M exact
  int e = (int)(i & 2047);
  size_t bm = i >> 11;
  int m = (int)(bm & 1023), b = (int)(bm >> 10);
  bf16 bv = (bf16)h[i];
  hbf[i] = bv;
  hbfT[((size_t)b * EN + e) * MTK + m] = bv;
}

// alpha_g[b,m,g] = sigmoid(h[b,m,:] . W_g[g,:] + b_g[g]); wave g per (b,m)
__global__ void nv_alphag(const float* __restrict__ h, const float* __restrict__ W_g,
                          const float* __restrict__ b_g, float* __restrict__ ag) {
  int bm = blockIdx.x;
  int g = threadIdx.x >> 5, lane = threadIdx.x & 31;
  const float* hr = h + (size_t)bm * EN;
  const float* wr = W_g + (size_t)g * EN;
  float s = 0.f;
  for (int c = lane; c < EN; c += 32) s += hr[c] * wr[c];
  for (int off = 16; off; off >>= 1) s += __shfl_down(s, off, 32);
  if (lane == 0) {
    float v = s + b_g[g];
    ag[(size_t)bm * GG + g] = 1.f / (1.f + __expf(-v));
  }
}

// softmax over M per (b,gk) column; fuse alpha_g; write w^T[b,g,k,m] bf16 + wsum
__global__ void nv_softmax_w(const float* __restrict__ logits, const float* __restrict__ ag,
                             bf16* __restrict__ wT, float* __restrict__ wsum) {
  int idx = blockIdx.x;                 // b*1024 + g*128 + k
  int b = idx >> 10, gk = idx & 1023, g = gk >> 7;
  const float* col = logits + (size_t)b * MTK * GKN + gk;     // stride GKN over m
  const float* agp = ag + (size_t)b * MTK * GG + g;           // stride GG over m
  __shared__ float red[256];
  int t = threadIdx.x;
  float mx = -1e30f;
  for (int m = t; m < MTK; m += 256) mx = fmaxf(mx, col[(size_t)m * GKN]);
  red[t] = mx; __syncthreads();
  for (int s = 128; s; s >>= 1) { if (t < s) red[t] = fmaxf(red[t], red[t + s]); __syncthreads(); }
  mx = red[0]; __syncthreads();
  float se = 0.f;
  for (int m = t; m < MTK; m += 256) se += __expf(col[(size_t)m * GKN] - mx);
  red[t] = se; __syncthreads();
  for (int s = 128; s; s >>= 1) { if (t < s) red[t] += red[t + s]; __syncthreads(); }
  float inv = 1.f / red[0]; __syncthreads();
  bf16* dst = wT + (size_t)idx * MTK;
  float ws = 0.f;
  for (int m = t; m < MTK; m += 256) {
    float wv = agp[(size_t)m * GG] * __expf(col[(size_t)m * GKN] - mx) * inv;
    dst[m] = (bf16)wv; ws += wv;
  }
  red[t] = ws; __syncthreads();
  for (int s = 128; s; s >>= 1) { if (t < s) red[t] += red[t + s]; __syncthreads(); }
  if (t == 0) wsum[idx] = red[0];
}

// vf then transpose: v[b,d,g] = sum_k vlad[b,g,k,d]*W_f[k] + b_f
__global__ void nv_vf(const float* __restrict__ vlad, const float* __restrict__ W_f,
                      const float* __restrict__ b_f, float* __restrict__ v) {
  int bg = blockIdx.x;                  // 32
  int b = bg >> 3, g = bg & 7, d = threadIdx.x;
  const float* base = vlad + (size_t)bg * KCL * DGD + d;
  float s = 0.f;
  for (int k = 0; k < KCL; ++k) s += base[(size_t)k * DGD] * W_f[k];
  v[((size_t)b * DGD + d) * GG + g] = s + b_f[0];
}

__global__ void nv_center(float* __restrict__ v) {      // center over last dim (8)
  int b = blockIdx.x, d = threadIdx.x;
  float* p = v + ((size_t)b * DGD + d) * GG;
  float mu = 0.f;
#pragma unroll
  for (int j = 0; j < 8; ++j) mu += p[j];
  mu *= 0.125f;
#pragma unroll
  for (int j = 0; j < 8; ++j) p[j] -= mu;
}

__global__ void nv_cov(const float* __restrict__ vc, float* __restrict__ cov) {
  int br = blockIdx.x;                  // b*256 + dr
  int b = br >> 8, dr = br & 255, dc = threadIdx.x;
  const float* pr = vc + ((size_t)b * DGD + dr) * GG;
  const float* pc = vc + ((size_t)b * DGD + dc) * GG;
  float s = 0.f;
#pragma unroll
  for (int j = 0; j < 8; ++j) s += pr[j] * pc[j];
  cov[((size_t)b * DD + dr) * DD + dc] = s * 0.125f;
}

__global__ void nv_trace(const float* __restrict__ cov, float* __restrict__ tr,
                         float* __restrict__ sq) {
  int b = blockIdx.x, t = threadIdx.x;
  __shared__ float red[256];
  red[t] = cov[((size_t)b * DD + t) * DD + t];
  __syncthreads();
  for (int s = 128; s; s >>= 1) { if (t < s) red[t] += red[t + s]; __syncthreads(); }
  if (t == 0) { tr[b] = red[0]; sq[b] = sqrtf(red[0]); }
}

__global__ void nv_nsinit(const float* __restrict__ cov, const float* __restrict__ tr,
                          float* __restrict__ A, float* __restrict__ ZY) {
  int br = blockIdx.x;                  // b*256 + r
  int b = br >> 8, r = br & 255, c = threadIdx.x;
  size_t idx = ((size_t)b * DD + r) * DD + c;
  float a = cov[idx] / tr[b];
  A[idx] = a;
  ZY[idx] = 0.5f * ((r == c ? 3.f : 0.f) - a);
}

// Batched 256x256 f32 matmul via v_wmma_f32_16x16x4_f32 with NS epilogues.
// mode 0: D=P   1: D=0.5*(3I-P)   2: D=3I-P   3: D=P*0.5*scale[b]
__global__ void nv_mm256(const float* __restrict__ A, const float* __restrict__ B,
                         float* __restrict__ D, const float* __restrict__ scale, int mode) {
  int wave = blockIdx.x * (blockDim.x >> 5) + (threadIdx.x >> 5);  // 1024 exact
  int lane = threadIdx.x & 31;
  int row = lane & 15, hf = lane >> 4;
  int b = wave >> 8;
  int r = wave & 255;
  int tm = r >> 4, tn = r & 15;
  const float* pa = A + ((size_t)b * DD + tm * 16 + row) * DD;      // A row, K contiguous
  const float* pb = B + (size_t)b * DD * DD + (tn * 16 + row);      // B column, stride DD
  f32x8 acc = {};
  for (int k = 0; k < DD; k += 4) {
    int kk = k + 2 * hf;
    f32x2 av = { pa[kk], pa[kk + 1] };
    f32x2 bv = { pb[(size_t)kk * DD], pb[(size_t)(kk + 1) * DD] };
    acc = __builtin_amdgcn_wmma_f32_16x16x4_f32(false, av, false, bv,
                                                (short)0, acc, false, false);
  }
  int col = tn * 16 + row;
  float* pd = D + ((size_t)b * DD + tm * 16 + hf * 8) * DD + col;
#pragma unroll
  for (int i = 0; i < 8; ++i) {
    int ro = tm * 16 + hf * 8 + i;
    float p = acc[i], v;
    if (mode == 0)      v = p;
    else if (mode == 1) v = 0.5f * ((ro == col ? 3.f : 0.f) - p);
    else if (mode == 2) v = (ro == col ? 3.f : 0.f) - p;
    else                v = p * 0.5f * scale[b];
    pd[(size_t)i * DD] = v;
  }
}

__global__ void nv_triuvec(const float* __restrict__ F, float* __restrict__ out) {
  int br = blockIdx.x;                  // b*256 + r
  int b = br >> 8, r = br & 255, c = threadIdx.x;
  if (c < r) return;
  size_t t = (size_t)r * DD - (size_t)r * (r - 1) / 2 + (c - r);
  out[(size_t)b * TRIU + t] = F[((size_t)b * DD + r) * DD + c];
}

// ---------------------------------------------------------------------------
extern "C" void kernel_launch(void* const* d_in, const int* in_sizes, int n_in,
                              void* d_out, int out_size, void* d_ws, size_t ws_size,
                              hipStream_t stream) {
  const float* x     = (const float*)d_in[0];
  const float* cent  = (const float*)d_in[1];
  const float* W_inp = (const float*)d_in[2];
  const float* b_inp = (const float*)d_in[3];
  const float* W_g   = (const float*)d_in[4];
  const float* b_g   = (const float*)d_in[5];
  const float* W_gk  = (const float*)d_in[6];
  const float* b_gk  = (const float*)d_in[7];
  const float* W_f   = (const float*)d_in[8];
  const float* b_f   = (const float*)d_in[9];
  float* out = (float*)d_out;

  char* ws = (char*)d_ws;
  size_t off = 0;
  auto take = [&](size_t bytes) -> char* {
    char* p = ws + off;
    off += (bytes + 255) & ~(size_t)255;
    return p;
  };
  bf16*  XTN   = (bf16*)take((size_t)BS * MTK * NF * 2);      // 8 MB
  bf16*  WINPB = (bf16*)take((size_t)EN * NF * 2);            // 4 MB
  bf16*  WGKB  = (bf16*)take((size_t)GKN * EN * 2);           // 4 MB
  float* H     = (float*)take((size_t)BS * MTK * EN * 4);     // 32 MB
  bf16*  HBF   = (bf16*)take((size_t)BS * MTK * EN * 2);      // 16 MB
  bf16*  HBFT  = (bf16*)take((size_t)BS * EN * MTK * 2);      // 16 MB
  float* LOGI  = (float*)take((size_t)BS * MTK * GKN * 4);    // 16 MB
  float* AG    = (float*)take((size_t)BS * MTK * GG * 4);
  bf16*  WT    = (bf16*)take((size_t)BS * GKN * MTK * 2);     // 8 MB
  float* WSUM  = (float*)take((size_t)BS * GKN * 4);
  float* VLAD  = (float*)take((size_t)BS * GG * KCL * DGD * 4);  // 4 MB
  float* VF    = (float*)take((size_t)BS * DGD * GG * 4);
  float* TR    = (float*)take(256);
  float* SQ    = (float*)take(256);
  float* COV   = (float*)take((size_t)BS * DD * DD * 4);
  float* ABUF  = (float*)take((size_t)BS * DD * DD * 4);
  float* ZY1   = (float*)take((size_t)BS * DD * DD * 4);
  float* Y1    = (float*)take((size_t)BS * DD * DD * 4);
  float* TB    = (float*)take((size_t)BS * DD * DD * 4);
  float* Y2    = (float*)take((size_t)BS * DD * DD * 4);
  float* Z2    = (float*)take((size_t)BS * DD * DD * 4);
  float* FB    = (float*)take((size_t)BS * DD * DD * 4);

  // 0) weight casts + input normalize
  nv_cast_bf16<<<(EN * NF + 255) / 256, 256, 0, stream>>>(W_inp, WINPB, EN * NF);
  nv_cast_bf16<<<(GKN * EN + 255) / 256, 256, 0, stream>>>(W_gk, WGKB, GKN * EN);
  nv_prep_normalize<<<BS * MTK, 256, 0, stream>>>(x, XTN);

  // 1) h = xt @ W_inp^T + b_inp   [4 x 1024 x 2048], K=1024  (TDM GEMM)
  nv_gemm_tdm<<<BS * 8 * 16, 256, 0, stream>>>(
      XTN, (long)MTK * NF, WINPB, 0, H, (long)MTK * EN,
      8, 16, NF, NF, NF, EN, b_inp, nullptr, nullptr, MTK, 0, BS);

  // 2) bf16 copies of h (row-major and transposed)
  nv_h_cast_tr<<<(BS * MTK * EN) / 256, 256, 0, stream>>>(H, HBF, HBFT);

  // 3) alpha_g
  nv_alphag<<<BS * MTK, 256, 0, stream>>>(H, W_g, b_g, AG);

  // 4) logits = h @ W_gk^T + b_gk   [4 x 1024 x 1024], K=2048  (TDM GEMM)
  nv_gemm_tdm<<<BS * 8 * 8, 256, 0, stream>>>(
      HBF, (long)MTK * EN, WGKB, 0, LOGI, (long)MTK * GKN,
      8, 8, EN, EN, EN, GKN, b_gk, nullptr, nullptr, MTK, 0, BS);

  // 5) softmax over M + alpha_g fuse -> w^T (bf16) + wsum
  nv_softmax_w<<<BS * GKN, 256, 0, stream>>>(LOGI, AG, WT, WSUM);

  // 6) vlad = w^T @ rx - wsum*centroids  (32 batches of [128 x 1024 x 256])
  //    HBFT batch stride: (b*EN + g*DGD)*MTK == bg*DGD*MTK since EN = 8*DGD.
  nv_gemm_tdm<<<BS * GG * 1 * 2, 256, 0, stream>>>(
      WT, (long)KCL * MTK, HBFT, (long)DGD * MTK, VLAD, (long)KCL * DGD,
      1, 2, MTK, MTK, MTK, DGD, nullptr, WSUM, cent, KCL, 1, BS * GG);

  // 7) cluster fusion + transpose + covpool
  nv_vf<<<BS * GG, DGD, 0, stream>>>(VLAD, W_f, b_f, VF);
  nv_center<<<BS, DGD, 0, stream>>>(VF);
  nv_cov<<<BS * DD, DD, 0, stream>>>(VF, COV);

  // 8) Newton-Schulz sqrtm (iter_n = 3), f32 WMMA matmuls
  nv_trace<<<BS, DD, 0, stream>>>(COV, TR, SQ);
  nv_nsinit<<<BS * DD, DD, 0, stream>>>(COV, TR, ABUF, ZY1);          // A, ZY=0.5(3I-A)
  nv_mm256<<<(BS * 256) / 8, 256, 0, stream>>>(ABUF, ZY1, Y1, SQ, 0); // Y1 = A@ZY ; Z1 = ZY1
  nv_mm256<<<(BS * 256) / 8, 256, 0, stream>>>(ZY1, Y1, TB, SQ, 1);   // T = 0.5(3I - Z1@Y1)
  nv_mm256<<<(BS * 256) / 8, 256, 0, stream>>>(Y1, TB, Y2, SQ, 0);    // Y2 = Y1@T
  nv_mm256<<<(BS * 256) / 8, 256, 0, stream>>>(TB, ZY1, Z2, SQ, 0);   // Z2 = T@Z1
  nv_mm256<<<(BS * 256) / 8, 256, 0, stream>>>(Z2, Y2, TB, SQ, 2);    // T = 3I - Z2@Y2
  nv_mm256<<<(BS * 256) / 8, 256, 0, stream>>>(Y2, TB, FB, SQ, 3);    // F = 0.5*sqrt(tr)*Y2@T

  // 9) upper-triangular vectorization -> output [4, 32896]
  nv_triuvec<<<BS * DD, DD, 0, stream>>>(FB, out);
}
